// GABBlock_54743653155458
// MI455X (gfx1250) — matrix operand order repaired and checked by reading, not compiled, not run on hardware
//
#include <hip/hip_runtime.h>
#include <hip/hip_bf16.h>

// ---------------------------------------------------------------------------
// CDNA5 / gfx1250 implementation of the GAB block.
// All GEMM-like work (2x conv3x3 implicit GEMM, dense projections, GAT
// flash-attention) goes through v_wmma_f32_16x16x32_f16 (f16 in, f32 acc).
// LDS staging of clean f16 tiles uses the CDNA5 async global->LDS engine
// (global_load_async_to_lds_b128 + s_wait_asynccnt), bypassing VGPRs.
// wave32: 128-thread blocks = 4 waves; each wave owns a 16-wide N strip.
// ---------------------------------------------------------------------------

typedef __attribute__((ext_vector_type(16))) _Float16 v16h;
typedef __attribute__((ext_vector_type(8)))  float    v8f;

static __device__ __forceinline__ v8f wmma_f16(v16h a, v16h b, v8f c) {
  // D = A(16x32) * B(32x16) + C(16x16), f32 accum
  return __builtin_amdgcn_wmma_f32_16x16x32_f16(false, a, false, b, (short)0, c,
                                                false, false);
}

// ---- CDNA5 async global->LDS copy (tracked by ASYNCcnt) --------------------
// Per ISA 08_async_tensor: each active lane copies 16B from its global
// address directly into LDS at VGPR[VDST] (wave-relative byte address).
// Generic LDS pointers carry the wave-relative offset in addr[31:0]
// (aperture rule: LDS address is the low 32 bits), so truncation is the address.
static __device__ __forceinline__ void async_ld_b128(void* lds, const void* gsrc) {
  const unsigned loff = (unsigned)(size_t)lds;
  asm volatile("global_load_async_to_lds_b128 %0, %1, off"
               :
               : "v"(loff), "v"(gsrc)
               : "memory");
}
static __device__ __forceinline__ void wait_asynccnt0() {
  asm volatile("s_wait_asynccnt 0x0" ::: "memory");
}

// A fragment (16x32 f16, MxK) in contiguous-k layout, from row-major [M][ld].
// lane l: M = l%16 ; lanes 0-15 hold K = 0..15, lanes 16-31 hold K = 16..31
// (contiguous, matching b_frag).
static __device__ __forceinline__ v16h a_frag(const _Float16* p, int ld) {
  const int l = threadIdx.x & 31;
  const _Float16* r = p + (l & 15) * ld + ((l >> 4) << 4);
  v16h a;
#pragma unroll
  for (int t = 0; t < 16; ++t) { a[t] = r[t]; }
  return a;
}

// B fragment (32x16 f16, KxN) from Bt layout [N][ld] (ld = K stride, mult of 16).
// lane l: N = l%16 ; lanes 0-15 hold K=0..15, lanes 16-31 hold K=16..31
// -> one contiguous 32B load per lane.
static __device__ __forceinline__ v16h b_frag(const _Float16* p, int ld) {
  const int l = threadIdx.x & 31;
  return *(const v16h*)(p + (l & 15) * ld + ((l >> 4) << 4));
}

// ---------------------------------------------------------------------------
// Weight prep
// ---------------------------------------------------------------------------

// conv weight (oc, ic, 3, 3) f32 -> f16 [oc][k] with k = (ky*3+kx)*64 + ic
__global__ void k_pack_convw(const float* __restrict__ w, _Float16* __restrict__ dst) {
  const int idx = blockIdx.x * 256 + threadIdx.x;
  if (idx >= 64 * 576) return;
  const int oc = idx / 576, k = idx % 576;
  const int kidx = k >> 6, ic = k & 63;
  const int ky = kidx / 3, kx = kidx % 3;
  dst[idx] = (_Float16)w[((oc * 64 + ic) * 3 + ky) * 3 + kx];
}

// src [K][N] f32 -> dst [N][K] f16
__global__ void k_transpose_f16(const float* __restrict__ src, _Float16* __restrict__ dst,
                                int K, int N) {
  const int idx = blockIdx.x * 256 + threadIdx.x;
  if (idx >= K * N) return;
  const int n = idx / K, k = idx % K;
  dst[idx] = (_Float16)src[k * N + n];
}

// ---------------------------------------------------------------------------
// Conv1 (3x3, C=64->64) + ReLU, implicit GEMM, writes window-token layout:
// tok[(i*64+j)*64 + n][c], n = (h%8)*8 + (w%8).
// grid (512, 32), block 128. Tile: 16 pixels x 64 oc, K = 576 in 18 chunks.
// (Staging needs f32->f16 conversion + halo predication, so it stays on the
//  VGPR path; the weights' B fragments are direct contiguous global loads.)
// ---------------------------------------------------------------------------
__global__ void k_conv1_relu(const float* __restrict__ x, const _Float16* __restrict__ w1t,
                             const float* __restrict__ cb, _Float16* __restrict__ tok) {
  __shared__ __align__(32) _Float16 As[16 * 32];
  const int tid = threadIdx.x, w = tid >> 5, l = tid & 31;
  const int h = blockIdx.x;
  const int w0 = blockIdx.y * 16;
  v8f acc = {};
  for (int kc = 0; kc < 18; ++kc) {
    const int kidx = kc >> 1, ic0 = (kc & 1) * 32;
    const int ky = kidx / 3, kx = kidx % 3;
    const int hh = h + ky - 1;
    __syncthreads();
#pragma unroll
    for (int e = 0; e < 4; ++e) {
      const int flat = tid + e * 128;
      const int m = flat & 15, t = flat >> 4;
      const int ww = w0 + m + kx - 1;
      float val = 0.f;
      if ((unsigned)hh < 512u && (unsigned)ww < 512u)
        val = x[(size_t)(ic0 + t) * (512 * 512) + hh * 512 + ww];
      As[m * 32 + t] = (_Float16)val;
    }
    __syncthreads();
    v16h af = a_frag(As, 32);
    v16h bf = b_frag(w1t + (size_t)(w * 16) * 576 + kc * 32, 576);
    acc = wmma_f16(af, bf, acc);
  }
  const int oc = w * 16 + (l & 15);
  const float bias = cb[oc];
#pragma unroll
  for (int v = 0; v < 8; ++v) {
    const int m = v + ((l >> 4) << 3);
    const int ww = w0 + m;
    float r = acc[v] + bias;
    r = r > 0.f ? r : 0.f;
    const int i = h >> 3, j = ww >> 3, n = (h & 7) * 8 + (ww & 7);
    tok[(((size_t)i * 64 + j) * 64 + n) * 64 + oc] = (_Float16)r;
  }
}

// ---------------------------------------------------------------------------
// DomainEmbedding, fused per window. grid 4096, block 128.
// KV = tok_win(64x64) @ dom_kv_w(64x128) + b   (WMMA in LDS)
// then 16 learned queries (4 heads x 4) cross-attend to 64 tokens (VALU).
// Node features written at l = j*64 + i, f = q*64 + h*16 + d (torch glue).
// ---------------------------------------------------------------------------
__global__ void k_domain_embed(const _Float16* __restrict__ tok,
                               const _Float16* __restrict__ dkv_t,
                               const float* __restrict__ dkv_b,
                               const float* __restrict__ dom_q,
                               _Float16* __restrict__ Xg) {
  __shared__ __align__(32) _Float16 Tw[64 * 64];
  __shared__ __align__(32) _Float16 KV[64 * 128];
  const int tid = threadIdx.x, w = tid >> 5, l = tid & 31;
  const int wi = blockIdx.x;
  const int i = wi >> 6, j = wi & 63;
  {  // async copy 8KB window straight into LDS (no VGPR round-trip)
    const _Float16* g = tok + (size_t)wi * 4096 + tid * 32;
    _Float16* s0 = &Tw[tid * 32];
#pragma unroll
    for (int e = 0; e < 4; ++e) async_ld_b128(s0 + e * 8, g + e * 8);
  }
  wait_asynccnt0();
  __syncthreads();
#pragma unroll
  for (int sp = 0; sp < 2; ++sp) {            // wave handles N strips w and w+4
    const int n0 = sp * 64 + w * 16;
    v8f acc[4] = {};
#pragma unroll
    for (int kc = 0; kc < 2; ++kc) {
      v16h bf = b_frag(dkv_t + (size_t)n0 * 64 + kc * 32, 64);
#pragma unroll
      for (int s = 0; s < 4; ++s) {
        v16h af = a_frag(&Tw[s * 16 * 64 + kc * 32], 64);
        acc[s] = wmma_f16(af, bf, acc[s]);
      }
    }
    const int nn = n0 + (l & 15);
    const float bias = dkv_b[nn];
#pragma unroll
    for (int s = 0; s < 4; ++s)
#pragma unroll
      for (int v = 0; v < 8; ++v) {
        const int m = s * 16 + v + ((l >> 4) << 3);
        KV[m * 128 + nn] = (_Float16)(acc[s][v] + bias);
      }
  }
  __syncthreads();
  if (tid < 16) {                              // one (head, query) per thread
    const int hh = tid >> 2, q = tid & 3;
    float dq[16];
#pragma unroll
    for (int d = 0; d < 16; ++d) dq[d] = dom_q[(hh * 4 + q) * 16 + d];
    float lg[64], mx = -1e30f;
    for (int n = 0; n < 64; ++n) {
      float s = 0.f;
#pragma unroll
      for (int d = 0; d < 16; ++d) s += dq[d] * (float)KV[n * 128 + hh * 16 + d];
      lg[n] = s;
      mx = fmaxf(mx, s);
    }
    float den = 0.f;
    for (int n = 0; n < 64; ++n) { lg[n] = __expf(lg[n] - mx); den += lg[n]; }
    const float inv = 1.f / den;
    const size_t node = (size_t)j * 64 + i;    // faithful glue transposition
#pragma unroll
    for (int d = 0; d < 16; ++d) {
      float o = 0.f;
      for (int n = 0; n < 64; ++n) o += lg[n] * (float)KV[n * 128 + 64 + hh * 16 + d];
      Xg[node * 256 + q * 64 + hh * 16 + d] = (_Float16)(o * inv);
    }
  }
}

// ---------------------------------------------------------------------------
// Generic WMMA GEMM: C[M][N] = A[M][K] @ Bt[N][K]^T + bias, f16 out.
// grid (M/64, N/64), block 128. K multiple of 32.
// A tile staged to LDS via the async engine.
// ---------------------------------------------------------------------------
__global__ void k_gemm_f16(const _Float16* __restrict__ A, const _Float16* __restrict__ Bt,
                           const float* __restrict__ bias, _Float16* __restrict__ Cm,
                           int M, int K, int N) {
  __shared__ __align__(32) _Float16 As[64 * 32];
  const int tid = threadIdx.x, w = tid >> 5, l = tid & 31;
  const int m0 = blockIdx.x * 64;
  const int n0 = blockIdx.y * 64 + w * 16;
  v8f acc[4] = {};
  const int nkc = K >> 5;
  for (int kc = 0; kc < nkc; ++kc) {
    __syncthreads();
    {  // async-stage 64x32 f16 A tile: two 16B lane-copies per thread
      const int row = tid >> 1, part = (tid & 1) * 16;
      const _Float16* g = A + (size_t)(m0 + row) * K + kc * 32 + part;
      _Float16* s0 = &As[row * 32 + part];
      async_ld_b128(s0, g);
      async_ld_b128(s0 + 8, g + 8);
    }
    wait_asynccnt0();
    __syncthreads();
    v16h bf = b_frag(Bt + (size_t)n0 * K + kc * 32, K);
#pragma unroll
    for (int s = 0; s < 4; ++s) {
      v16h af = a_frag(&As[s * 16 * 32], 32);
      acc[s] = wmma_f16(af, bf, acc[s]);
    }
  }
  const int nn = n0 + (l & 15);
  const float bv = bias ? bias[nn] : 0.f;
#pragma unroll
  for (int s = 0; s < 4; ++s)
#pragma unroll
    for (int v = 0; v < 8; ++v) {
      const int mm = m0 + s * 16 + v + ((l >> 4) << 3);
      Cm[(size_t)mm * N + nn] = (_Float16)(acc[s][v] + bv);
    }
}

// ---------------------------------------------------------------------------
// GAT attention, flash style (no 1/sqrt(d) scale, matching the reference).
// grid (64 row-tiles, 4 heads), block 128; wave owns 16 query rows.
// S = Q@K^T via WMMA; online softmax with 16-lane shfl reductions (C-layout
// row = v + 8*(lane>=16) keeps per-row stats in-lane); P staged per-wave in
// LDS; V staged transposed so PV B-fragments are contiguous. Next K/V chunk
// prefetched into L2 (global_prefetch_b8).
// ---------------------------------------------------------------------------
__global__ void k_gat_flash(const _Float16* __restrict__ QKV, _Float16* __restrict__ Og) {
  __shared__ __align__(32) _Float16 Vt[64 * 64];        // [d][key]
  __shared__ __align__(32) _Float16 Pl[4][16 * 64];     // per-wave P
  const int tid = threadIdx.x, w = tid >> 5, l = tid & 31;
  const int h = blockIdx.y;
  const int qbase = blockIdx.x * 64 + w * 16;
  const _Float16* qh = QKV + h * 64;
  const _Float16* kh = QKV + 256 + h * 64;
  const _Float16* vh = QKV + 512 + h * 64;
  v16h aq0 = a_frag(qh + (size_t)qbase * 768, 768);
  v16h aq1 = a_frag(qh + (size_t)qbase * 768 + 32, 768);
  v8f o[4] = {};
  float mrun[8], lrun[8];
#pragma unroll
  for (int v = 0; v < 8; ++v) { mrun[v] = -1e30f; lrun[v] = 0.f; }
  for (int kb = 0; kb < 4096; kb += 64) {
    if (kb + 64 < 4096) {  // L2 prefetch of the next streamed chunk
      const int key = tid >> 1;
      __builtin_prefetch(kh + (size_t)(kb + 64 + key) * 768, 0, 1);
      __builtin_prefetch(vh + (size_t)(kb + 64 + key) * 768, 0, 1);
    }
    __syncthreads();
    {  // stage V chunk transposed
      const int key = tid >> 1, d0 = (tid & 1) * 32;
      const _Float16* src = vh + (size_t)(kb + key) * 768 + d0;
#pragma unroll
      for (int q = 0; q < 32; ++q) Vt[(d0 + q) * 64 + key] = src[q];
    }
    __syncthreads();
    v8f s[4] = {};
#pragma unroll
    for (int nt = 0; nt < 4; ++nt) {
      const _Float16* bp = kh + (size_t)(kb + nt * 16) * 768;
      s[nt] = wmma_f16(aq0, b_frag(bp, 768), s[nt]);
      s[nt] = wmma_f16(aq1, b_frag(bp + 32, 768), s[nt]);
    }
    float alpha[8];
#pragma unroll
    for (int v = 0; v < 8; ++v) {
      float mx = fmaxf(fmaxf(s[0][v], s[1][v]), fmaxf(s[2][v], s[3][v]));
#pragma unroll
      for (int off = 1; off < 16; off <<= 1) mx = fmaxf(mx, __shfl_xor(mx, off, 32));
      const float mnew = fmaxf(mrun[v], mx);
      alpha[v] = __expf(mrun[v] - mnew);
      float ps = 0.f;
#pragma unroll
      for (int nt = 0; nt < 4; ++nt) {
        const float p = __expf(s[nt][v] - mnew);
        s[nt][v] = p;
        ps += p;
      }
#pragma unroll
      for (int off = 1; off < 16; off <<= 1) ps += __shfl_xor(ps, off, 32);
      lrun[v] = lrun[v] * alpha[v] + ps;
      mrun[v] = mnew;
    }
#pragma unroll
    for (int nt = 0; nt < 4; ++nt)
#pragma unroll
      for (int v = 0; v < 8; ++v) {
        const int row = v + ((l >> 4) << 3);
        Pl[w][row * 64 + nt * 16 + (l & 15)] = (_Float16)s[nt][v];
        o[nt][v] *= alpha[v];
      }
    v16h ap0 = a_frag(&Pl[w][0], 64);
    v16h ap1 = a_frag(&Pl[w][32], 64);
#pragma unroll
    for (int nt = 0; nt < 4; ++nt) {
      o[nt] = wmma_f16(ap0, b_frag(&Vt[nt * 16 * 64], 64), o[nt]);
      o[nt] = wmma_f16(ap1, b_frag(&Vt[nt * 16 * 64 + 32], 64), o[nt]);
    }
  }
#pragma unroll
  for (int nt = 0; nt < 4; ++nt)
#pragma unroll
    for (int v = 0; v < 8; ++v) {
      const int row = qbase + v + ((l >> 4) << 3);
      const int d = nt * 16 + (l & 15);
      Og[(size_t)row * 256 + h * 64 + d] = (_Float16)(o[nt][v] / lrun[v]);
    }
}

// ---------------------------------------------------------------------------
// Intra-domain attention, fused per window. grid 4096, block 128.
// Q2 = tok_win @ intra_q_w (WMMA); global tokens from O2 row i*64+j with
// feature reinterpretation f = c*QS + qs; tiny 4-key attention in VALU;
// writes conv2 input image (f16, NCHW).
// ---------------------------------------------------------------------------
__global__ void k_intra(const _Float16* __restrict__ tok,
                        const _Float16* __restrict__ iq_t, const float* __restrict__ iq_b,
                        const _Float16* __restrict__ ikv_t, const float* __restrict__ ikv_b,
                        const _Float16* __restrict__ O2, _Float16* __restrict__ yimg) {
  __shared__ __align__(32) _Float16 Tw[64 * 64];
  __shared__ __align__(32) _Float16 Q2[64 * 64];
  __shared__ float Gtf[4 * 64];
  __shared__ float KV2[4 * 128];
  const int tid = threadIdx.x, w = tid >> 5, l = tid & 31;
  const int wi = blockIdx.x;
  const int i = wi >> 6, j = wi & 63;
  {  // async copy 8KB window straight into LDS
    const _Float16* g = tok + (size_t)wi * 4096 + tid * 32;
    _Float16* s0 = &Tw[tid * 32];
#pragma unroll
    for (int e = 0; e < 4; ++e) async_ld_b128(s0 + e * 8, g + e * 8);
  }
#pragma unroll
  for (int rep = 0; rep < 2; ++rep) {          // gt[qs][c] = O2[wi][c*4+qs]
    const int e = tid + rep * 128;
    const int qs = e >> 6, c = e & 63;
    Gtf[e] = (float)O2[(size_t)wi * 256 + c * 4 + qs];
  }
  wait_asynccnt0();
  __syncthreads();
  {  // Q2 = Tw @ intra_q_w + b
    const int n0 = w * 16;
    v8f acc[4] = {};
#pragma unroll
    for (int kc = 0; kc < 2; ++kc) {
      v16h bf = b_frag(iq_t + (size_t)n0 * 64 + kc * 32, 64);
#pragma unroll
      for (int s = 0; s < 4; ++s) {
        v16h af = a_frag(&Tw[s * 16 * 64 + kc * 32], 64);
        acc[s] = wmma_f16(af, bf, acc[s]);
      }
    }
    const int nn = n0 + (l & 15);
    const float bias = iq_b[nn];
#pragma unroll
    for (int s = 0; s < 4; ++s)
#pragma unroll
      for (int v = 0; v < 8; ++v) {
        const int m = s * 16 + v + ((l >> 4) << 3);
        Q2[m * 64 + nn] = (_Float16)(acc[s][v] + bias);
      }
  }
  {  // KV2 = Gt(4x64) @ intra_kv_w(64x128) + b   (tiny -> VALU)
    const int col = tid;
    const float b0 = ikv_b[col];
    float accm[4] = {b0, b0, b0, b0};
    for (int k = 0; k < 64; ++k) {
      const float wv = (float)ikv_t[(size_t)col * 64 + k];
#pragma unroll
      for (int m = 0; m < 4; ++m) accm[m] += Gtf[m * 64 + k] * wv;
    }
#pragma unroll
    for (int m = 0; m < 4; ++m) KV2[m * 128 + col] = accm[m];
  }
  __syncthreads();
#pragma unroll
  for (int rep = 0; rep < 2; ++rep) {          // (token, head) pairs
    const int p = tid + rep * 128;
    const int n = p >> 2, hh = p & 3;
    float lg[4], mx = -1e30f;
#pragma unroll
    for (int m = 0; m < 4; ++m) {
      float s = 0.f;
#pragma unroll
      for (int d = 0; d < 16; ++d)
        s += (float)Q2[n * 64 + hh * 16 + d] * KV2[m * 128 + hh * 16 + d];
      lg[m] = s;
      mx = fmaxf(mx, s);
    }
    float den = 0.f;
#pragma unroll
    for (int m = 0; m < 4; ++m) { lg[m] = __expf(lg[m] - mx); den += lg[m]; }
    const float inv = 1.f / den;
    const int hrow = i * 8 + (n >> 3), wcol = j * 8 + (n & 7);
#pragma unroll
    for (int d = 0; d < 16; ++d) {
      float o = 0.f;
#pragma unroll
      for (int m = 0; m < 4; ++m) o += lg[m] * KV2[m * 128 + 64 + hh * 16 + d];
      const int c = hh * 16 + d;
      yimg[((size_t)c * 512 + hrow) * 512 + wcol] = (_Float16)(o * inv);
    }
  }
}

// ---------------------------------------------------------------------------
// Conv2 (3x3) + bias + residual, implicit GEMM from f16 image, f32 out.
// ---------------------------------------------------------------------------
__global__ void k_conv2_res(const _Float16* __restrict__ yimg, const _Float16* __restrict__ w2t,
                            const float* __restrict__ cb, const float* __restrict__ x,
                            float* __restrict__ out) {
  __shared__ __align__(32) _Float16 As[16 * 32];
  const int tid = threadIdx.x, w = tid >> 5, l = tid & 31;
  const int h = blockIdx.x;
  const int w0 = blockIdx.y * 16;
  v8f acc = {};
  for (int kc = 0; kc < 18; ++kc) {
    const int kidx = kc >> 1, ic0 = (kc & 1) * 32;
    const int ky = kidx / 3, kx = kidx % 3;
    const int hh = h + ky - 1;
    __syncthreads();
#pragma unroll
    for (int e = 0; e < 4; ++e) {
      const int flat = tid + e * 128;
      const int m = flat & 15, t = flat >> 4;
      const int ww = w0 + m + kx - 1;
      _Float16 val = (_Float16)0.f;
      if ((unsigned)hh < 512u && (unsigned)ww < 512u)
        val = yimg[(size_t)(ic0 + t) * (512 * 512) + hh * 512 + ww];
      As[m * 32 + t] = val;
    }
    __syncthreads();
    v16h af = a_frag(As, 32);
    v16h bf = b_frag(w2t + (size_t)(w * 16) * 576 + kc * 32, 576);
    acc = wmma_f16(af, bf, acc);
  }
  const int oc = w * 16 + (l & 15);
  const float bias = cb[oc];
#pragma unroll
  for (int v = 0; v < 8; ++v) {
    const int m = v + ((l >> 4) << 3);
    const int ww = w0 + m;
    const size_t idx = ((size_t)oc * 512 + h) * 512 + ww;
    out[idx] = acc[v] + bias + x[idx];
  }
}

// ---------------------------------------------------------------------------
// Host side
// ---------------------------------------------------------------------------
extern "C" void kernel_launch(void* const* d_in, const int* in_sizes, int n_in,
                              void* d_out, int out_size, void* d_ws, size_t ws_size,
                              hipStream_t stream) {
  (void)in_sizes; (void)n_in; (void)out_size; (void)ws_size;
  const float* x          = (const float*)d_in[0];
  const float* conv_w     = (const float*)d_in[1];
  const float* conv_b     = (const float*)d_in[2];
  const float* dom_q      = (const float*)d_in[3];
  const float* dom_kv_w   = (const float*)d_in[4];
  const float* dom_kv_b   = (const float*)d_in[5];
  const float* intra_q_w  = (const float*)d_in[6];
  const float* intra_q_b  = (const float*)d_in[7];
  const float* intra_kv_w = (const float*)d_in[8];
  const float* intra_kv_b = (const float*)d_in[9];
  const float* gat_qkv_w  = (const float*)d_in[10];
  const float* gat_qkv_b  = (const float*)d_in[11];
  const float* gat_out_w  = (const float*)d_in[12];
  const float* gat_out_b  = (const float*)d_in[13];
  const float* out_w      = (const float*)d_in[14];
  const float* out_b      = (const float*)d_in[15];
  float* outF = (float*)d_out;

  char* ws = (char*)d_ws;
  auto carve = [&](size_t bytes) -> void* {
    void* p = (void*)ws;
    ws += (bytes + 255) & ~(size_t)255;
    return p;
  };
  _Float16* tok   = (_Float16*)carve((size_t)4096 * 64 * 64 * 2);   // 33.5 MB
  _Float16* Xg    = (_Float16*)carve((size_t)4096 * 256 * 2);       //  2.1 MB
  _Float16* QKV   = (_Float16*)carve((size_t)4096 * 768 * 2);       //  6.3 MB
  _Float16* Og    = (_Float16*)carve((size_t)4096 * 256 * 2);
  _Float16* O2    = (_Float16*)carve((size_t)4096 * 256 * 2);
  _Float16* Yimg  = (_Float16*)carve((size_t)64 * 512 * 512 * 2);   // 33.5 MB
  _Float16* w1t   = (_Float16*)carve((size_t)64 * 576 * 2);
  _Float16* w2t   = (_Float16*)carve((size_t)64 * 576 * 2);
  _Float16* dkvT  = (_Float16*)carve((size_t)128 * 64 * 2);
  _Float16* iqT   = (_Float16*)carve((size_t)64 * 64 * 2);
  _Float16* ikvT  = (_Float16*)carve((size_t)128 * 64 * 2);
  _Float16* gqkvT = (_Float16*)carve((size_t)768 * 256 * 2);
  _Float16* goutT = (_Float16*)carve((size_t)256 * 256 * 2);

  // weight prep (f32 -> f16, B^T layout for contiguous B fragments)
  k_pack_convw<<<144, 256, 0, stream>>>(conv_w, w1t);
  k_pack_convw<<<144, 256, 0, stream>>>(out_w, w2t);
  k_transpose_f16<<<(64 * 128 + 255) / 256, 256, 0, stream>>>(dom_kv_w, dkvT, 64, 128);
  k_transpose_f16<<<(64 * 64 + 255) / 256, 256, 0, stream>>>(intra_q_w, iqT, 64, 64);
  k_transpose_f16<<<(64 * 128 + 255) / 256, 256, 0, stream>>>(intra_kv_w, ikvT, 64, 128);
  k_transpose_f16<<<(256 * 768 + 255) / 256, 256, 0, stream>>>(gat_qkv_w, gqkvT, 256, 768);
  k_transpose_f16<<<(256 * 256 + 255) / 256, 256, 0, stream>>>(gat_out_w, goutT, 256, 256);

  // pipeline
  k_conv1_relu<<<dim3(512, 32), 128, 0, stream>>>(x, w1t, conv_b, tok);
  k_domain_embed<<<4096, 128, 0, stream>>>(tok, dkvT, dom_kv_b, dom_q, Xg);
  k_gemm_f16<<<dim3(64, 12), 128, 0, stream>>>(Xg, gqkvT, gat_qkv_b, QKV, 4096, 256, 768);
  k_gat_flash<<<dim3(64, 4), 128, 0, stream>>>(QKV, Og);
  k_gemm_f16<<<dim3(64, 4), 128, 0, stream>>>(Og, goutT, gat_out_b, O2, 4096, 256, 256);
  k_intra<<<4096, 128, 0, stream>>>(tok, iqT, intra_q_b, ikvT, intra_kv_b, O2, Yimg);
  k_conv2_res<<<dim3(512, 32), 128, 0, stream>>>(Yimg, w2t, out_b, x, outF);
}
